// ODEBlock_6468220748711
// MI455X (gfx1250) — matrix operand order repaired and checked by reading, not compiled, not run on hardware
//
#include <hip/hip_runtime.h>

// Problem constants from the reference
#define B_DIM 512
#define D_DIM 1024
#define H_DIM 4096
#define T_PTS 32
#define DT    0.05f

typedef __attribute__((ext_vector_type(16))) __bf16 v16bf;
typedef __attribute__((ext_vector_type(8)))  float  v8f;

// 32-byte fragment: 16 bf16 values viewed as 2x uint4 for b128 loads
union Frag {
    v16bf v;
    uint4 q[2];
    unsigned short u[16];
};

__device__ __forceinline__ unsigned short f2bf(float f) {
    // round-to-nearest-even fp32 -> bf16
    unsigned u = __float_as_uint(f);
    u += 0x7FFFu + ((u >> 16) & 1u);
    return (unsigned short)(u >> 16);
}

// ---------------------------------------------------------------------------
// Transpose-convert: in fp32 [R][C] row-major -> out bf16 [C][R] row-major.
// Gives weights the [N][K] layout so WMMA B-fragments are contiguous loads.
// ---------------------------------------------------------------------------
__global__ void k_transpose_bf16(const float* __restrict__ in,
                                 unsigned short* __restrict__ out,
                                 int R, int C) {
    __shared__ float tile[32][33];
    const int c0 = blockIdx.x * 32, r0 = blockIdx.y * 32;
    const int tx = threadIdx.x, ty = threadIdx.y;
#pragma unroll
    for (int i = 0; i < 4; ++i)
        tile[ty + i * 8][tx] = in[(size_t)(r0 + ty + i * 8) * C + (c0 + tx)];
    __syncthreads();
#pragma unroll
    for (int i = 0; i < 4; ++i)
        out[(size_t)(c0 + ty + i * 8) * R + (r0 + tx)] = f2bf(tile[tx][ty + i * 8]);
}

// out[t=0] slab = x
__global__ void k_copy(const float* __restrict__ in, float* __restrict__ out, int n) {
    int i = blockIdx.x * blockDim.x + threadIdx.x;
    if (i < n) out[i] = in[i];
}

// Stage input: yin_bf16 = bf16(y + alpha * k_prev)   (alpha=0 / k=null for stage 1)
__global__ void k_prep_a(const float* __restrict__ y, const float* __restrict__ kadd,
                         float alpha, unsigned short* __restrict__ out, int n) {
    int i = blockIdx.x * blockDim.x + threadIdx.x;
    if (i >= n) return;
    float v = y[i];
    if (kadd) v += alpha * kadd[i];
    out[i] = f2bf(v);
}

// y_{i+1} = y_i + h/6 (k1 + 2 k2 + 2 k3 + k4)
__global__ void k_rk4_combine(const float* __restrict__ y0,
                              const float* __restrict__ k1, const float* __restrict__ k2,
                              const float* __restrict__ k3, const float* __restrict__ k4,
                              float h, float* __restrict__ y1, int n) {
    int i = blockIdx.x * blockDim.x + threadIdx.x;
    if (i >= n) return;
    y1[i] = y0[i] + (h * (1.0f / 6.0f)) * (k1[i] + 2.0f * k2[i] + 2.0f * k3[i] + k4[i]);
}

// ---------------------------------------------------------------------------
// WMMA GEMM:  C[M,N] = A[M,K](bf16, row-major) * Bt[N,K](bf16)^T + bias[N]
// mode 1: C = tanh(...) stored as bf16 (feeds next GEMM's A operand)
// mode 0: C stored fp32 (RK stage derivative)
// Block: 256 threads = 8 waves; waves 2(M) x 4(N); wave tile 32x64 (8 WMMAs).
// Block tile 64 x 256.  Grid: (N/256, M/64).  K multiple of 32.
// ---------------------------------------------------------------------------
__global__ void __launch_bounds__(256)
k_gemm_wmma(const unsigned short* __restrict__ A,
            const unsigned short* __restrict__ Bt,
            const float* __restrict__ bias,
            float* __restrict__ Cf,
            unsigned short* __restrict__ Cb,
            int M, int N, int K, int mode) {
    const int lane = threadIdx.x & 31;
    const int wave = threadIdx.x >> 5;
    const int wm   = wave >> 2;                 // 0..1
    const int wn   = wave & 3;                  // 0..3
    const int row0 = blockIdx.y * 64 + wm * 32;
    const int col0 = blockIdx.x * 256 + wn * 64;
    const int r    = lane & 15;                 // row (A) / col (B,C) within tile
    const int hh   = lane >> 4;                 // lane-half selector

    v8f acc[2][4] = {};

    // Per-row / per-col base pointers in uint4 (8 bf16) units
    const uint4* a_ptr[2];
#pragma unroll
    for (int mt = 0; mt < 2; ++mt)
        a_ptr[mt] = (const uint4*)(A + (size_t)(row0 + mt * 16 + r) * K);
    const uint4* b_ptr[4];
#pragma unroll
    for (int nt = 0; nt < 4; ++nt)
        b_ptr[nt] = (const uint4*)(Bt + (size_t)(col0 + nt * 16 + r) * K);

    for (int k0 = 0; k0 < K; k0 += 32) {
        const int ka = (k0 >> 3) + hh;      // A chunk0: K = k0 + hh*8 .. +7
        const int kb = (k0 >> 3) + hh;      // B chunk0: K = k0 + hh*8 .. +7
        Frag a[2], b[4];
#pragma unroll
        for (int mt = 0; mt < 2; ++mt) {
            a[mt].q[0] = a_ptr[mt][ka];       // e0..7  : K = k0 + hh*8 + e
            a[mt].q[1] = a_ptr[mt][ka + 2];   // e8..15 : K = k0 + 16 + hh*8 + (e-8)
        }
#pragma unroll
        for (int nt = 0; nt < 4; ++nt) {
            b[nt].q[0] = b_ptr[nt][kb];       // e0..7  : K = k0 + hh*8 + e
            b[nt].q[1] = b_ptr[nt][kb + 2];   // e8..15 : K = k0 + 16 + hh*8 + (e-8)
        }
        if (k0 + 32 < K)                      // hint next A chunk toward L2/L0
            __builtin_prefetch((const void*)(a_ptr[0] + ka + 4), 0, 1);
#pragma unroll
        for (int mt = 0; mt < 2; ++mt)
#pragma unroll
            for (int nt = 0; nt < 4; ++nt)
                acc[mt][nt] = __builtin_amdgcn_wmma_f32_16x16x32_bf16(
                    false, a[mt].v, false, b[nt].v,
                    (short)0, acc[mt][nt], false, false);
    }

    // Epilogue: C/D layout — lane half hh, col = r, VGPR v -> row = v + 8*hh
#pragma unroll
    for (int mt = 0; mt < 2; ++mt) {
#pragma unroll
        for (int nt = 0; nt < 4; ++nt) {
            const int col = col0 + nt * 16 + r;
            const float bv = bias[col];
#pragma unroll
            for (int v = 0; v < 8; ++v) {
                const int row = row0 + mt * 16 + v + 8 * hh;
                float val = acc[mt][nt][v] + bv;
                if (mode) {
                    val = tanhf(val);
                    Cb[(size_t)row * N + col] = f2bf(val);
                } else {
                    Cf[(size_t)row * N + col] = val;
                }
            }
        }
    }
}

// ---------------------------------------------------------------------------
extern "C" void kernel_launch(void* const* d_in, const int* in_sizes, int n_in,
                              void* d_out, int out_size, void* d_ws, size_t ws_size,
                              hipStream_t stream) {
    const float* x  = (const float*)d_in[0];   // [B,D]
    const float* W1 = (const float*)d_in[1];   // [D,H]
    const float* b1 = (const float*)d_in[2];   // [H]
    const float* W2 = (const float*)d_in[3];   // [H,D]
    const float* b2 = (const float*)d_in[4];   // [D]
    float* out = (float*)d_out;                // [T,B,D]

    // Workspace carve-up (~29 MB)
    char* ws = (char*)d_ws;
    unsigned short* w1t = (unsigned short*)ws;                                 // [H][D] bf16, 8MB
    unsigned short* w2t = (unsigned short*)(ws + (size_t)D_DIM * H_DIM * 2);   // [D][H] bf16, 8MB
    unsigned short* yin = (unsigned short*)(ws + (size_t)D_DIM * H_DIM * 4);   // [B,D] bf16, 1MB
    unsigned short* hb  = yin + (size_t)B_DIM * D_DIM;                         // [B,H] bf16, 4MB
    float* kb = (float*)(hb + (size_t)B_DIM * H_DIM);                          // 4 x [B,D] fp32
    const int n_bd = B_DIM * D_DIM;
    float* ks[4] = { kb, kb + n_bd, kb + 2 * n_bd, kb + 3 * n_bd };

    // Weight transpose+convert (bf16 [N][K] layout for contiguous B-fragments)
    dim3 tb_t(32, 8);
    k_transpose_bf16<<<dim3(H_DIM / 32, D_DIM / 32), tb_t, 0, stream>>>(W1, w1t, D_DIM, H_DIM);
    k_transpose_bf16<<<dim3(D_DIM / 32, H_DIM / 32), tb_t, 0, stream>>>(W2, w2t, H_DIM, D_DIM);

    // out[t=0] = x
    k_copy<<<n_bd / 256, 256, 0, stream>>>(x, out, n_bd);

    const dim3 g1(H_DIM / 256, B_DIM / 64);   // GEMM1: 512x4096x1024
    const dim3 g2(D_DIM / 256, B_DIM / 64);   // GEMM2: 512x1024x4096

    for (int i = 0; i < T_PTS - 1; ++i) {
        const float* y0 = out + (size_t)i * n_bd;
        const float* kprev = nullptr;
        float alpha = 0.0f;
        for (int j = 0; j < 4; ++j) {
            // Yin = y0 + alpha * k_{j-1} (bf16)
            k_prep_a<<<n_bd / 256, 256, 0, stream>>>(y0, kprev, alpha, yin, n_bd);
            // H = tanh(Yin * W1 + b1)  -> bf16
            k_gemm_wmma<<<g1, 256, 0, stream>>>(yin, w1t, b1, nullptr, hb,
                                                B_DIM, H_DIM, D_DIM, 1);
            // k_j = H * W2 + b2        -> fp32
            k_gemm_wmma<<<g2, 256, 0, stream>>>(hb, w2t, b2, ks[j], nullptr,
                                                B_DIM, D_DIM, H_DIM, 0);
            kprev = ks[j];
            alpha = (j < 2) ? (DT * 0.5f) : DT;
        }
        // y_{i+1} = y_i + dt/6 (k1 + 2k2 + 2k3 + k4), written to next output slab
        k_rk4_combine<<<n_bd / 256, 256, 0, stream>>>(y0, ks[0], ks[1], ks[2], ks[3],
                                                      DT, out + (size_t)(i + 1) * n_bd, n_bd);
    }
}
